// CustomCell_30691836297631
// MI455X (gfx1250) — matrix operand order repaired and checked by reading, not compiled, not run on hardware
//
#include <hip/hip_runtime.h>
#include <hip/hip_bf16.h>

typedef __attribute__((ext_vector_type(16))) __bf16 v16bf;
typedef __attribute__((ext_vector_type(8)))  __bf16 v8bf;
typedef __attribute__((ext_vector_type(8)))  float  v8f;

#define HH 512
#define DD 256
#define VV 33
#define TT 256
#define KK 512

// ---------------- WMMA fragment loaders (CDNA5 16x16x32 bf16 layouts) ----------------

// A: 16x32 (MxK) bf16 from row-major buffer (LDS). Lane<16: row=lane, K 0-7 & 16-23.
// Lane>=16: row=lane-16, K 8-15 & 24-31.  (vector elems 0..7 = K half*8.., 8..15 = K 16+half*8..)
__device__ __forceinline__ v16bf load_a_frag(const __bf16* base, int stride, int kbase, int lane)
{
    const int row  = lane & 15;
    const int half = lane >> 4;
    const __bf16* p = base + row * stride + kbase + half * 8;
    v8bf lo = *(const v8bf*)(p);
    v8bf hi = *(const v8bf*)(p + 16);
    v16bf r;
#pragma unroll
    for (int i = 0; i < 8; ++i) { r[i] = lo[i]; r[i + 8] = hi[i]; }
    return r;
}

// B: 32x16 (KxN) bf16, sourced from transposed weights WT[N][Kdim] (row-major).
// Lane n: N = n&15, holds K = (n>>4)*8 .. +7 and 16+(n>>4)*8 .. +7 (same order as A).
__device__ __forceinline__ v16bf load_b_frag(const __bf16* bt, int Kdim, int ncol0, int kbase, int lane)
{
    const __bf16* p = bt + (size_t)(ncol0 + (lane & 15)) * Kdim + kbase + (lane >> 4) * 8;
    v8bf lo = *(const v8bf*)(p);
    v8bf hi = *(const v8bf*)(p + 16);
    v16bf r;
#pragma unroll
    for (int i = 0; i < 8; ++i) { r[i] = lo[i]; r[i + 8] = hi[i]; }
    return r;
}

__device__ __forceinline__ v8f wmma_bf16(v16bf a, v16bf b, v8f c)
{
    return __builtin_amdgcn_wmma_f32_16x16x32_bf16(false, a, false, b, (short)0, c,
                                                   false, false);
}

// Single-accumulator GEMM segment (attention query, output dense).
__device__ __forceinline__ void gemm_seg(v8f& acc,
                                         const __bf16* A, int strideA, int kLen,
                                         const __bf16* BT, int Kdim, int ncol0, int kOffB,
                                         int lane)
{
#pragma unroll 2
    for (int k = 0; k < kLen; k += 32) {
        v16bf a = load_a_frag(A, strideA, k, lane);
        v16bf b = load_b_frag(BT, Kdim, ncol0, kOffB + k, lane);
        acc = wmma_bf16(a, b, acc);
    }
}

// Fused 4-gate GEMM segment: one A fragment feeds the i/f/g/o column tiles
// (cols j, 512+j, 1024+j, 1536+j) -> 4 WMMAs per A load, 4x less LDS traffic.
__device__ __forceinline__ void gemm_seg4(v8f& zi, v8f& zf, v8f& zg, v8f& zo,
                                          const __bf16* A, int strideA, int kLen,
                                          const __bf16* BT, int Kdim, int j, int kOffB,
                                          int lane)
{
#pragma unroll 1
    for (int k = 0; k < kLen; k += 32) {
        v16bf a  = load_a_frag(A, strideA, k, lane);
        v16bf b0 = load_b_frag(BT, Kdim, j,        kOffB + k, lane);
        v16bf b1 = load_b_frag(BT, Kdim, 512 + j,  kOffB + k, lane);
        v16bf b2 = load_b_frag(BT, Kdim, 1024 + j, kOffB + k, lane);
        v16bf b3 = load_b_frag(BT, Kdim, 1536 + j, kOffB + k, lane);
        zi = wmma_bf16(a, b0, zi);
        zf = wmma_bf16(a, b1, zf);
        zg = wmma_bf16(a, b2, zg);
        zo = wmma_bf16(a, b3, zo);
    }
}

__device__ __forceinline__ float sigmoidf_(float v) { return 1.f / (1.f + __expf(-v)); }

// ---------------- LSTM layer: z = [seg0|a] @ W + h_own @ U + b, gates i,f,g,o ----------------
// Wave w owns output cols [w*64, w*64+64) of each 512-wide gate; 4 j-tiles of 16.
// New h goes to hnew (LDS) so the jt loop stays rolled and register pressure stays low;
// h is published with a cheap LDS->LDS copy after the barrier.
__device__ __forceinline__ void lstm_layer(
    const __bf16* seg0, int seg0Stride, int seg0Len,
    const __bf16* abuf,
    const __bf16* hown,
    const __bf16* WT, int WKdim,
    const __bf16* UT,
    const float* bias, float* cbuf, __bf16* hbuf, __bf16* hnew,
    int lane, int wave, int tid)
{
    const int half = lane >> 4;
#pragma unroll 1
    for (int jt = 0; jt < 4; ++jt) {
        const int j = wave * 64 + jt * 16;
        v8f zi = {0.f,0.f,0.f,0.f,0.f,0.f,0.f,0.f};
        v8f zf = zi, zg = zi, zo = zi;
        // x-part (or h_prev layer) | attention part | recurrent h @ U
        gemm_seg4(zi, zf, zg, zo, seg0, seg0Stride, seg0Len, WT, WKdim, j, 0,       lane);
        gemm_seg4(zi, zf, zg, zo, abuf, DD,         DD,      WT, WKdim, j, seg0Len, lane);
        gemm_seg4(zi, zf, zg, zo, hown, HH,         HH,      UT, HH,    j, 0,       lane);

        const int col = j + (lane & 15);
        const float bi = bias[col], bf_ = bias[512 + col];
        const float bg = bias[1024 + col], bo_ = bias[1536 + col];
#pragma unroll
        for (int r = 0; r < 8; ++r) {
            const int row = r + half * 8;
            const float ig = sigmoidf_(zi[r] + bi);
            const float fg = sigmoidf_(zf[r] + bf_);
            const float og = sigmoidf_(zo[r] + bo_);
            const float gg = tanhf(zg[r] + bg);
            const float cn = fg * cbuf[row * HH + col] + ig * gg;  // c is wave-private by col
            cbuf[row * HH + col] = cn;
            hnew[row * HH + col] = (__bf16)(og * tanhf(cn));       // hnew never read by GEMMs
        }
    }
    __syncthreads();   // all waves done READING hown/h (and writing hnew)
    for (int i = tid; i < 16 * HH; i += 256) hbuf[i] = hnew[i];
    __syncthreads();
}

// ---------------- Luong attention: q = h@Wq+bq (WMMA); scores/softmax/read = VALU ----------------
__device__ __forceinline__ void attention(
    const __bf16* hsrc, const __bf16* WqT, const float* bq,
    const float* __restrict__ keys, int rowBase,
    float* qbuf, float* wbuf, __bf16* abuf,
    int lane, int wave)
{
    const int half = lane >> 4;
#pragma unroll 1
    for (int ti = 0; ti < 2; ++ti) {
        const int n0 = wave * 32 + ti * 16;
        v8f acc = {0.f,0.f,0.f,0.f,0.f,0.f,0.f,0.f};
        gemm_seg(acc, hsrc, HH, HH, WqT, HH, n0, 0, lane);
        const int col = n0 + (lane & 15);
        const float bb = bq[col];
#pragma unroll
        for (int r = 0; r < 8; ++r)
            qbuf[(r + half * 8) * DD + col] = acc[r] + bb;
    }
    __syncthreads();

    // each wave handles 2 batch rows; 512 scores, 16 per lane (k = i*32+lane)
#pragma unroll 1
    for (int rr = 0; rr < 2; ++rr) {
        const int r = wave * 2 + rr;
        const float* krow = keys + (size_t)(rowBase + r) * (KK * DD);
        const float4* qp = (const float4*)(qbuf + r * DD);
        float s[16];
#pragma unroll
        for (int i = 0; i < 16; ++i) {
            const int k = i * 32 + lane;
            const float4* kp = (const float4*)(krow + (size_t)k * DD);
            float acc = 0.f;
#pragma unroll 4
            for (int d = 0; d < 64; ++d) {
                float4 kv = kp[d], qv = qp[d];
                acc += kv.x * qv.x + kv.y * qv.y + kv.z * qv.z + kv.w * qv.w;
            }
            s[i] = acc;
        }
        float mx = s[0];
#pragma unroll
        for (int i = 1; i < 16; ++i) mx = fmaxf(mx, s[i]);
#pragma unroll
        for (int off = 16; off >= 1; off >>= 1) mx = fmaxf(mx, __shfl_xor(mx, off, 32));
        float sum = 0.f;
#pragma unroll
        for (int i = 0; i < 16; ++i) { s[i] = __expf(s[i] - mx); sum += s[i]; }
#pragma unroll
        for (int off = 16; off >= 1; off >>= 1) sum += __shfl_xor(sum, off, 32);
        const float inv = 1.f / sum;
#pragma unroll
        for (int i = 0; i < 16; ++i) wbuf[r * KK + i * 32 + lane] = s[i] * inv;
    }
    __syncthreads();

    // read = softmax . keys ; each lane owns 8 consecutive d values
#pragma unroll 1
    for (int rr = 0; rr < 2; ++rr) {
        const int r = wave * 2 + rr;
        const float* krow = keys + (size_t)(rowBase + r) * (KK * DD);
        const int d0 = lane * 8;
        float a0 = 0.f, a1 = 0.f, a2 = 0.f, a3 = 0.f, a4 = 0.f, a5 = 0.f, a6 = 0.f, a7 = 0.f;
#pragma unroll 4
        for (int k = 0; k < KK; ++k) {
            const float wk = wbuf[r * KK + k];
            const float4* kp = (const float4*)(krow + (size_t)k * DD + d0);
            float4 v0 = kp[0], v1 = kp[1];
            a0 += wk * v0.x; a1 += wk * v0.y; a2 += wk * v0.z; a3 += wk * v0.w;
            a4 += wk * v1.x; a5 += wk * v1.y; a6 += wk * v1.z; a7 += wk * v1.w;
        }
        abuf[r * DD + d0 + 0] = (__bf16)a0; abuf[r * DD + d0 + 1] = (__bf16)a1;
        abuf[r * DD + d0 + 2] = (__bf16)a2; abuf[r * DD + d0 + 3] = (__bf16)a3;
        abuf[r * DD + d0 + 4] = (__bf16)a4; abuf[r * DD + d0 + 5] = (__bf16)a5;
        abuf[r * DD + d0 + 6] = (__bf16)a6; abuf[r * DD + d0 + 7] = (__bf16)a7;
    }
    __syncthreads();
}

// ---------------- main persistent decoder kernel: 1 WG = 16 batch rows, loops T ----------------
__global__ void __launch_bounds__(256, 1)
decoder_kernel(const float* __restrict__ x, const float* __restrict__ keys,
               const __bf16* __restrict__ W1t, const __bf16* __restrict__ U1t, const float* __restrict__ b1,
               const __bf16* __restrict__ W2t, const __bf16* __restrict__ U2t, const float* __restrict__ b2,
               const __bf16* __restrict__ W3t, const __bf16* __restrict__ U3t, const float* __restrict__ b3,
               const __bf16* __restrict__ Wqt, const float* __restrict__ bq,
               const __bf16* __restrict__ Wot, const float* __restrict__ bo,
               float* __restrict__ out)
{
    __shared__ __align__(32) __bf16 h1[16 * HH], h2[16 * HH], h3[16 * HH];   // 48 KB
    __shared__ __align__(32) float  c1[16 * HH], c2[16 * HH], c3[16 * HH];   // 96 KB
    __shared__ __align__(32) __bf16 hnew[16 * HH];                           // 16 KB
    __shared__ __align__(32) __bf16 abuf[16 * DD];                           // 8 KB
    __shared__ __align__(32) __bf16 xin[16 * 64];                            // 2 KB
    __shared__ __align__(16) float  qbuf[16 * DD];                           // 16 KB
    __shared__ __align__(16) float  wbuf[16 * KK];                           // 32 KB

    const int tid     = threadIdx.x;
    const int lane    = tid & 31;
    const int wave    = tid >> 5;
    const int rowBase = blockIdx.x * 16;

    for (int i = tid; i < 16 * HH; i += 256) {
        h1[i] = (__bf16)0.f; h2[i] = (__bf16)0.f; h3[i] = (__bf16)0.f;
        c1[i] = 0.f; c2[i] = 0.f; c3[i] = 0.f;
    }
    for (int i = tid; i < 16 * DD; i += 256) abuf[i] = (__bf16)0.f;
    __syncthreads();

    for (int t = 0; t < TT; ++t) {
        // stage x_t (pad 33 -> 64 with zeros, matching zero-padded W1t cols)
        for (int i = tid; i < 16 * 64; i += 256) {
            const int r = i >> 6, v = i & 63;
            float val = (v < VV) ? x[((size_t)(rowBase + r) * TT + t) * VV + v] : 0.f;
            xin[i] = (__bf16)val;
        }
        __syncthreads();

        lstm_layer(xin, 64, 64, abuf, h1, W1t, 320, U1t, b1, c1, h1, hnew, lane, wave, tid);
        attention(h1, Wqt, bq, keys, rowBase, qbuf, wbuf, abuf, lane, wave);
        lstm_layer(h1, HH, HH, abuf, h2, W2t, 768, U2t, b2, c2, h2, hnew, lane, wave, tid);
        attention(h2, Wqt, bq, keys, rowBase, qbuf, wbuf, abuf, lane, wave);
        lstm_layer(h2, HH, HH, abuf, h3, W3t, 768, U3t, b3, c3, h3, hnew, lane, wave, tid);
        attention(h3, Wqt, bq, keys, rowBase, qbuf, wbuf, abuf, lane, wave);

        // out = concat(h3, a3) @ Wo + bo ; 3 N-tiles (vocab padded 33 -> 48)
        if (wave < 3) {
            const int n0 = wave * 16;
            v8f acc = {0.f,0.f,0.f,0.f,0.f,0.f,0.f,0.f};
            gemm_seg(acc, h3, HH, HH, Wot, 768, n0, 0, lane);
            gemm_seg(acc, abuf, DD, DD, Wot, 768, n0, HH, lane);
            const int col  = n0 + (lane & 15);
            const int half = lane >> 4;
            if (col < VV) {
                const float bb = bo[col];
#pragma unroll
                for (int r = 0; r < 8; ++r)
                    out[((size_t)(rowBase + r + half * 8) * TT + t) * VV + col] = acc[r] + bb;
            }
        }
        __syncthreads();
    }
}

// ---------------- prep: zero + transpose/cast fp32 weights -> bf16 W^T in workspace ----------------
__global__ void ws_zero_kernel(__bf16* p, size_t n)
{
    size_t i = (size_t)blockIdx.x * blockDim.x + threadIdx.x;
    if (i < n) p[i] = (__bf16)0.f;
}

// dst[n][dstK0 + k] = src[srcRow0 + k][n]   (dst row-major with row length dstK)
__global__ void tcopy_kernel(__bf16* dst, int dstK, const float* __restrict__ src, int srcN,
                             int srcRow0, int dstK0, int copyK, int copyN)
{
    long idx = (long)blockIdx.x * blockDim.x + threadIdx.x;
    long total = (long)copyK * copyN;
    if (idx >= total) return;
    int n = (int)(idx / copyK);
    int k = (int)(idx % copyK);
    dst[(size_t)n * dstK + dstK0 + k] = (__bf16)src[(size_t)(srcRow0 + k) * srcN + n];
}

extern "C" void kernel_launch(void* const* d_in, const int* in_sizes, int n_in,
                              void* d_out, int out_size, void* d_ws, size_t ws_size,
                              hipStream_t stream)
{
    const float* x    = (const float*)d_in[0];
    const float* keys = (const float*)d_in[1];
    const float* W1   = (const float*)d_in[2];
    const float* U1   = (const float*)d_in[3];
    const float* b1   = (const float*)d_in[4];
    const float* W2   = (const float*)d_in[5];
    const float* U2   = (const float*)d_in[6];
    const float* b2   = (const float*)d_in[7];
    const float* W3   = (const float*)d_in[8];
    const float* U3   = (const float*)d_in[9];
    const float* b3   = (const float*)d_in[10];
    const float* Wq   = (const float*)d_in[11];
    const float* bq   = (const float*)d_in[12];
    const float* Wo   = (const float*)d_in[13];
    const float* bo   = (const float*)d_in[14];
    float* out = (float*)d_out;

    __bf16* ws = (__bf16*)d_ws;
    // element offsets in workspace
    const size_t oW1 = 0;                              // [2048][320]  (x pad 64 | a 256)
    const size_t oW2 = oW1 + (size_t)2048 * 320;       // [2048][768]
    const size_t oW3 = oW2 + (size_t)2048 * 768;       // [2048][768]
    const size_t oU1 = oW3 + (size_t)2048 * 768;       // [2048][512]
    const size_t oU2 = oU1 + (size_t)2048 * 512;
    const size_t oU3 = oU2 + (size_t)2048 * 512;
    const size_t oWq = oU3 + (size_t)2048 * 512;       // [256][512]
    const size_t oWo = oWq + (size_t)256 * 512;        // [48][768]
    const size_t total = oWo + (size_t)48 * 768;
    if (ws_size < total * sizeof(__bf16)) return;

    ws_zero_kernel<<<(unsigned)((total + 255) / 256), 256, 0, stream>>>(ws, total);

    auto tc = [&](size_t dstOff, int dstK, const float* src, int srcN,
                  int srcRow0, int dstK0, int copyK, int copyN) {
        long n = (long)copyK * copyN;
        tcopy_kernel<<<(unsigned)((n + 255) / 256), 256, 0, stream>>>(
            ws + dstOff, dstK, src, srcN, srcRow0, dstK0, copyK, copyN);
    };
    tc(oW1, 320, W1, 2048,  0,  0,  33, 2048);   // x part of W1 (rows 0..32) -> cols 0..32
    tc(oW1, 320, W1, 2048, 33, 64, 256, 2048);   // a part of W1 (rows 33..288) -> cols 64..319
    tc(oW2, 768, W2, 2048,  0,  0, 768, 2048);
    tc(oW3, 768, W3, 2048,  0,  0, 768, 2048);
    tc(oU1, 512, U1, 2048,  0,  0, 512, 2048);
    tc(oU2, 512, U2, 2048,  0,  0, 512, 2048);
    tc(oU3, 512, U3, 2048,  0,  0, 512, 2048);
    tc(oWq, 512, Wq,  256,  0,  0, 512,  256);
    tc(oWo, 768, Wo,   33,  0,  0, 768,   33);

    decoder_kernel<<<16, 256, 0, stream>>>(x, keys,
        ws + oW1, ws + oU1, b1,
        ws + oW2, ws + oU2, b2,
        ws + oW3, ws + oU3, b3,
        ws + oWq, bq, ws + oWo, bo, out);
}
